// MHCN_78941498901027
// MI455X (gfx1250) — compile-verified, matrix operand order checked by reading, not run on hardware
//
#include <hip/hip_runtime.h>
#include <hip/hip_bf16.h>
#include <math.h>

typedef __attribute__((ext_vector_type(16))) __bf16 v16bf;
typedef __attribute__((ext_vector_type(8)))  float  v8f;

#define kN1 300000
#define kE1 400000
#define kN2 75000
#define kE2 100000
#define kN3 20000
#define kE3 40000
#define kG  1024
#define kD  64

// ---- ISA 16-bit A/B fragment index: lane holds 16 bf16; K index within a 32-block.
__device__ __forceinline__ int ak_idx(int lane, int i) {
  int kb = (lane & 16) ? 8 : 0;
  int v = i >> 1, h = i & 1;
  return (v < 4) ? (kb + 2 * v + h) : (16 + kb + 2 * (v - 4) + h);
}

__device__ __forceinline__ float elu1(float v) { return v > 0.f ? v : (expf(v) - 1.f); }

// ------------------------------------------------------------------ utilities
__global__ void fillz(float* __restrict__ p, size_t n) {
  size_t i = (size_t)blockIdx.x * blockDim.x + threadIdx.x;
  if (i < n) p[i] = 0.f;
}

// Pack NNConv edge-net weights (be folded in as extra K rows) directly into WMMA
// B-fragment order: out[(((b*4 + t)*32 + lane)*16 + i)] = W[k][n], k=b*32+ak_idx(lane,i),
// n=t*16+(lane&15).  One v16bf (32B) per (block,tile,lane) -> contiguous vector load.
__global__ void pack_Wext_swz(const float* __restrict__ We, const float* __restrict__ be,
                              __bf16* __restrict__ out, int fe, int fi, int nblk) {
  size_t id = (size_t)blockIdx.x * blockDim.x + threadIdx.x;
  size_t total = (size_t)nblk * 4 * 32 * 16;
  if (id >= total) return;
  int i = (int)(id & 15);
  int l = (int)((id >> 4) & 31);
  int t = (int)((id >> 9) & 3);
  int b = (int)(id >> 11);
  int k = b * 32 + ak_idx(l, i);
  int o = t * 16 + (l & 15);
  int KI = fe * fi;
  float v = 0.f;
  if (k < KI)           v = We[(size_t)k * 64 + o];
  else if (k < KI + fi) v = be[(size_t)(k - KI) * 64 + o];
  out[id] = (__bf16)v;
}

// Same fragment-order packing for a dense [K x N] weight, NT tiles of 16 columns.
__global__ void pack_W_swz(const float* __restrict__ W, __bf16* __restrict__ out,
                           int K, int N, int nblk, int NT) {
  size_t id = (size_t)blockIdx.x * blockDim.x + threadIdx.x;
  size_t total = (size_t)nblk * NT * 32 * 16;
  if (id >= total) return;
  int i = (int)(id & 15);
  int l = (int)((id >> 4) & 31);
  int rest = (int)(id >> 9);
  int t = rest % NT;
  int b = rest / NT;
  int k = b * 32 + ak_idx(l, i);
  int n = t * 16 + (l & 15);
  float v = (k < K && n < N) ? W[(size_t)k * N + n] : 0.f;
  out[id] = (__bf16)v;
}

// sum[idst[j]][f] += X[(isrc? isrc[j]:j)][f]; cnt[idst[j]] += 1 (f==0)
__global__ void seg_scatter_add(const float* __restrict__ X, int ldx,
                                const int* __restrict__ isrc, const int* __restrict__ idst,
                                float* __restrict__ sum, int lds, float* __restrict__ cnt,
                                int n, int F) {
  size_t id = (size_t)blockIdx.x * blockDim.x + threadIdx.x;
  if (id >= (size_t)n * F) return;
  int j = (int)(id / F);
  int f = (int)(id % F);
  int s = isrc ? isrc[j] : j;
  int d = idst[j];
  atomicAdd(&sum[(size_t)d * lds + f], X[(size_t)s * ldx + f]);
  if (cnt && f == 0) atomicAdd(&cnt[d], 1.f);
}

__global__ void div_cnt(float* __restrict__ sum, int ld, const float* __restrict__ cnt,
                        int n, int F) {
  size_t id = (size_t)blockIdx.x * blockDim.x + threadIdx.x;
  if (id >= (size_t)n * F) return;
  int i = (int)(id / F);
  int f = (int)(id % F);
  float c = cnt[i];
  sum[(size_t)i * ld + f] /= (c > 1.f ? c : 1.f);
}

__global__ void copy_cols(const float* __restrict__ src, int lds,
                          float* __restrict__ dst, int ldd, int col0, int n, int F) {
  size_t id = (size_t)blockIdx.x * blockDim.x + threadIdx.x;
  if (id >= (size_t)n * F) return;
  int i = (int)(id / F);
  int f = (int)(id % F);
  dst[(size_t)i * ldd + col0 + f] = src[(size_t)i * lds + f];
}

__global__ void log_softmax10(const float* __restrict__ in, int ld, float* __restrict__ out, int n) {
  int r = blockIdx.x * blockDim.x + threadIdx.x;
  if (r >= n) return;
  float m = -INFINITY;
  for (int c = 0; c < 10; ++c) m = fmaxf(m, in[(size_t)r * ld + c]);
  float s = 0.f;
  for (int c = 0; c < 10; ++c) s += expf(in[(size_t)r * ld + c] - m);
  float lse = m + logf(s);
  for (int c = 0; c < 10; ++c) out[(size_t)r * 10 + c] = in[(size_t)r * ld + c] - lse;
}

// ------------------------------------------------------ NNConv edge messages
// One wave per 16-edge tile.  z[e][k] = ea[e][k>>LFI]*xs[e][k&(FI-1)] for k<FE*FI,
// xs[e][k-FE*FI] for the folded-in `be` rows.  FE*FI is a multiple of 32, so the K loop
// splits into two straight-line phases (no divergent loads).  B fragments are
// pre-swizzled -> single v16bf vector load per fragment.
template <int FE, int FI, int LFI, int KPAD>
__global__ void __launch_bounds__(256)
nnconv_msg_wmma(const float* __restrict__ x, const int* __restrict__ ei,
                const float* __restrict__ ea, const __bf16* __restrict__ Wswz,
                float* __restrict__ agg, int nE) {
  constexpr int KI = FE * FI;       // multiple of 32 for all configs used
  __shared__ float shx[8][16][64];
  __shared__ float shea[8][16][16];
  __shared__ int   shd[8][16];
  int wave = threadIdx.x >> 5, lane = threadIdx.x & 31;
  int e0 = (blockIdx.x * 8 + wave) * 16;

  for (int t = lane; t < 16 * FI; t += 32) {
    int r = t >> LFI, c = t & (FI - 1);
    int e = e0 + r;
    float v = 0.f;
    if (e < nE) v = x[(size_t)ei[e] * FI + c];   // gather x[src]
    shx[wave][r][c] = v;
  }
  for (int t = lane; t < 16 * FE; t += 32) {
    int r = t / FE, c = t - r * FE;
    int e = e0 + r;
    shea[wave][r][c] = (e < nE) ? ea[(size_t)e * FE + c] : 0.f;
  }
  if (lane < 16) {
    int e = e0 + lane;
    shd[wave][lane] = (e < nE) ? ei[(size_t)nE + e] : -1;
  }
  __syncthreads();

  v8f acc[4];
#pragma unroll
  for (int t = 0; t < 4; ++t)
#pragma unroll
    for (int q = 0; q < 8; ++q) acc[t][q] = 0.f;

  int row = lane & 15;
  int akv[16];
#pragma unroll
  for (int i = 0; i < 16; ++i) akv[i] = ak_idx(lane, i);

  const v16bf* Wf = (const v16bf*)Wswz;

  // phase 1: k in [0, KI) -> ea * xs, unconditional LDS loads
  for (int kk = 0; kk < KI; kk += 32) {
    v16bf a;
#pragma unroll
    for (int i = 0; i < 16; ++i) {
      int k = kk + akv[i];
      a[i] = (__bf16)(shea[wave][row][k >> LFI] * shx[wave][row][k & (FI - 1)]);
    }
    size_t fb = (size_t)(kk >> 5) * 4 * 32 + lane;
#pragma unroll
    for (int t = 0; t < 4; ++t) {
      v16bf b = Wf[fb + (size_t)t * 32];
      acc[t] = __builtin_amdgcn_wmma_f32_16x16x32_bf16(false, a, false, b,
                                                       (short)0, acc[t], false, false);
    }
  }
  // phase 2: k in [KI, KPAD) -> xs (bias rows), zero pad past FI
#pragma unroll
  for (int kk = KI; kk < KPAD; kk += 32) {
    v16bf a;
#pragma unroll
    for (int i = 0; i < 16; ++i) {
      int kb = kk - KI + akv[i];
      if (FI >= 32) {
        a[i] = (__bf16)shx[wave][row][kb & (FI - 1)];
      } else {
        float v = shx[wave][row][kb & (FI - 1)];
        a[i] = (__bf16)(kb < FI ? v : 0.f);
      }
    }
    size_t fb = (size_t)(kk >> 5) * 4 * 32 + lane;
#pragma unroll
    for (int t = 0; t < 4; ++t) {
      v16bf b = Wf[fb + (size_t)t * 32];
      acc[t] = __builtin_amdgcn_wmma_f32_16x16x32_bf16(false, a, false, b,
                                                       (short)0, acc[t], false, false);
    }
  }

  int col = lane & 15;
  int mh = (lane & 16) ? 8 : 0;
#pragma unroll
  for (int r = 0; r < 8; ++r) {
    int d = shd[wave][r + mh];
    if (d >= 0) {
#pragma unroll
      for (int t = 0; t < 4; ++t)
        atomicAdd(&agg[(size_t)d * 64 + t * 16 + col], acc[t][r]);
    }
  }
}

// --------------------------------------------- generic Y = act(X@W + b + addend)
// W pre-swizzled into fragment order, NT tiles of 16 cols.  X loads are clamped
// in-bounds and value-selected (no predicated loads).
template <int NT>
__global__ void __launch_bounds__(256)
wmma_linear(const float* __restrict__ X, int ldx,
            const __bf16* __restrict__ Wswz, const float* __restrict__ bias,
            const float* __restrict__ addend, int lda,
            float* __restrict__ Y, int ldy,
            int M, int K, int Kpad, int Nv, int act) {
  int wave = threadIdx.x >> 5, lane = threadIdx.x & 31;
  int r0 = (blockIdx.x * 8 + wave) * 16;
  int row = lane & 15, col = lane & 15;

  v8f acc[NT];
#pragma unroll
  for (int t = 0; t < NT; ++t)
#pragma unroll
    for (int q = 0; q < 8; ++q) acc[t][q] = 0.f;

  int akv[16];
#pragma unroll
  for (int i = 0; i < 16; ++i) akv[i] = ak_idx(lane, i);

  const v16bf* Wf = (const v16bf*)Wswz;
  int rr = r0 + row;
  int rrc = rr < M ? rr : (M - 1);
  const float* Xr = X + (size_t)rrc * ldx;
  bool rowok = rr < M;

  for (int kk = 0; kk < Kpad; kk += 32) {
    v16bf a;
#pragma unroll
    for (int i = 0; i < 16; ++i) {
      int k = kk + akv[i];
      int kc = k < K ? k : (K - 1);
      float v = Xr[kc];
      a[i] = (__bf16)((rowok && k < K) ? v : 0.f);
    }
    size_t fb = ((size_t)(kk >> 5) * NT) * 32 + lane;
#pragma unroll
    for (int t = 0; t < NT; ++t) {
      v16bf b = Wf[fb + (size_t)t * 32];
      acc[t] = __builtin_amdgcn_wmma_f32_16x16x32_bf16(false, a, false, b,
                                                       (short)0, acc[t], false, false);
    }
  }

  int mh = (lane & 16) ? 8 : 0;
#pragma unroll
  for (int r = 0; r < 8; ++r) {
    int m = r0 + r + mh;
    if (m < M) {
#pragma unroll
      for (int t = 0; t < NT; ++t) {
        int n = t * 16 + col;
        float v = acc[t][r];
        if (bias && n < Nv) v += bias[n];
        if (addend) v += addend[(size_t)m * lda + n];
        if (act) v = elu1(v);
        Y[(size_t)m * ldy + n] = v;
      }
    }
  }
}

// ---------------------------------------------------------------- host driver
static inline int divup(int a, int b) { return (a + b - 1) / b; }

extern "C" void kernel_launch(void* const* d_in, const int* in_sizes, int n_in,
                              void* d_out, int out_size, void* d_ws, size_t ws_size,
                              hipStream_t stream) {
  (void)in_sizes; (void)n_in; (void)out_size; (void)ws_size;

  const float* in_x   = (const float*)d_in[0];
  const int*   ei1    = (const int*)d_in[1];
  const float* ea1    = (const float*)d_in[2];
  const int*   batch1 = (const int*)d_in[3];
  const int*   as12   = (const int*)d_in[4];
  const float* aedge2 = (const float*)d_in[5];
  const int*   ei2    = (const int*)d_in[6];
  const float* ea2    = (const float*)d_in[7];
  const int*   batch2 = (const int*)d_in[8];
  const int*   as23   = (const int*)d_in[9];
  const float* aedge3 = (const float*)d_in[10];
  const int*   ei3    = (const int*)d_in[11];
  const float* ea3    = (const float*)d_in[12];
  const int*   batch3 = (const int*)d_in[13];
  const int H11 = 14, H12 = 18, H13 = 22, V1 = 26, H21 = 28, H22 = 32, V2 = 36,
            H31 = 38, H32 = 42, FC1 = 46, FC2 = 48, FC3 = 50;
  float* out = (float*)d_out;

  char* base = (char*)d_ws;
  size_t off = 0;
  auto alloc = [&](size_t elems, size_t esz) -> void* {
    off = (off + 255) & ~(size_t)255;
    void* p = base + off;
    off += elems * esz;
    return p;
  };

  float* bufA = (float*)alloc((size_t)kN1 * 64, 4);
  float* bufB = (float*)alloc((size_t)kN1 * 64, 4);
  float* agg  = (float*)alloc((size_t)kN1 * 64, 4);
  float* x1g  = (float*)alloc((size_t)kG * 64, 4);
  float* x2g  = (float*)alloc((size_t)kG * 64, 4);
  float* x3g  = (float*)alloc((size_t)kG * 64, 4);
  float* cntG = (float*)alloc(kG, 4);
  float* cat2 = (float*)alloc((size_t)kN2 * 68, 4);
  float* cnt2 = (float*)alloc(kN2, 4);
  float* x2a  = (float*)alloc((size_t)kN2 * 64, 4);
  float* x2b  = (float*)alloc((size_t)kN2 * 64, 4);
  float* cat3 = (float*)alloc((size_t)kN3 * 80, 4);
  float* cnt3 = (float*)alloc(kN3, 4);
  float* x3a  = (float*)alloc((size_t)kN3 * 64, 4);
  float* x3b  = (float*)alloc((size_t)kN3 * 64, 4);
  float* xcat = (float*)alloc((size_t)kG * 192, 4);
  float* g1   = (float*)alloc((size_t)kG * 64, 4);
  float* g2   = (float*)alloc((size_t)kG * 32, 4);
  float* g3   = (float*)alloc((size_t)kG * 16, 4);

  // swizzled bf16 weights: nblk * NT * 512 elements each
  __bf16* w11e = (__bf16*)alloc((size_t)3 * 4 * 512, 2);    // Kpad 96
  __bf16* w12e = (__bf16*)alloc((size_t)10 * 4 * 512, 2);   // Kpad 320
  __bf16* w13e = (__bf16*)alloc((size_t)10 * 4 * 512, 2);
  __bf16* w21e = (__bf16*)alloc((size_t)34 * 4 * 512, 2);   // Kpad 1088
  __bf16* w22e = (__bf16*)alloc((size_t)34 * 4 * 512, 2);
  __bf16* w31e = (__bf16*)alloc((size_t)34 * 4 * 512, 2);
  __bf16* w32e = (__bf16*)alloc((size_t)34 * 4 * 512, 2);
  __bf16* w11r = (__bf16*)alloc((size_t)1 * 4 * 512, 2);    // K16 -> Kpad 32
  __bf16* w12r = (__bf16*)alloc((size_t)2 * 4 * 512, 2);    // K64
  __bf16* w13r = (__bf16*)alloc((size_t)2 * 4 * 512, 2);
  __bf16* w21r = (__bf16*)alloc((size_t)2 * 4 * 512, 2);
  __bf16* w22r = (__bf16*)alloc((size_t)2 * 4 * 512, 2);
  __bf16* w31r = (__bf16*)alloc((size_t)2 * 4 * 512, 2);
  __bf16* w32r = (__bf16*)alloc((size_t)2 * 4 * 512, 2);
  __bf16* wv1  = (__bf16*)alloc((size_t)3 * 4 * 512, 2);    // K68 -> 96
  __bf16* wv2  = (__bf16*)alloc((size_t)3 * 4 * 512, 2);    // K80 -> 96
  __bf16* wfc1 = (__bf16*)alloc((size_t)6 * 4 * 512, 2);    // K192
  __bf16* wfc2 = (__bf16*)alloc((size_t)2 * 2 * 512, 2);    // K64, NT2
  __bf16* wfc3 = (__bf16*)alloc((size_t)1 * 1 * 512, 2);    // K32, NT1

  auto FILL = [&](float* p, size_t n) {
    fillz<<<divup((int)n, 256), 256, 0, stream>>>(p, n);
  };
  auto PEXT = [&](int idx, __bf16* dst, int fe, int fi, int nblk) {
    int tot = nblk * 4 * 512;
    pack_Wext_swz<<<divup(tot, 256), 256, 0, stream>>>(
        (const float*)d_in[idx], (const float*)d_in[idx + 1], dst, fe, fi, nblk);
  };
  auto PW = [&](int idx, __bf16* dst, int K, int N, int nblk, int NT) {
    int tot = nblk * NT * 512;
    pack_W_swz<<<divup(tot, 256), 256, 0, stream>>>((const float*)d_in[idx], dst, K, N, nblk, NT);
  };
  auto SCAT = [&](const float* X, int ldx, const int* is, const int* id, float* s, int lds,
                  float* c, int n, int F) {
    seg_scatter_add<<<divup(n * F, 256), 256, 0, stream>>>(X, ldx, is, id, s, lds, c, n, F);
  };
  auto DIV = [&](float* s, int ld, const float* c, int n, int F) {
    div_cnt<<<divup(n * F, 256), 256, 0, stream>>>(s, ld, c, n, F);
  };
  auto CPY = [&](const float* s, int lds, float* d, int ldd, int c0, int n, int F) {
    copy_cols<<<divup(n * F, 256), 256, 0, stream>>>(s, lds, d, ldd, c0, n, F);
  };

#define MSG_BLK(nE) divup(divup((nE), 16), 8)
#define LIN_BLK(M)  divup(divup((M), 16), 8)
#define LIN4(X, ldx, W, b, add, lda, Y, ldy, M, K, Kp, Nv, act) \
  wmma_linear<4><<<LIN_BLK(M), 256, 0, stream>>>(X, ldx, W, b, add, lda, Y, ldy, M, K, Kp, Nv, act)

  // ---- pack weights (bf16, fragment-swizzled)
  PEXT(H11, w11e, 4, 16, 3);
  PEXT(H12, w12e, 4, 64, 10);
  PEXT(H13, w13e, 4, 64, 10);
  PEXT(H21, w21e, 16, 64, 34);
  PEXT(H22, w22e, 16, 64, 34);
  PEXT(H31, w31e, 16, 64, 34);
  PEXT(H32, w32e, 16, 64, 34);
  PW(H11 + 2, w11r, 16, 64, 1, 4);
  PW(H12 + 2, w12r, 64, 64, 2, 4);
  PW(H13 + 2, w13r, 64, 64, 2, 4);
  PW(H21 + 2, w21r, 64, 64, 2, 4);
  PW(H22 + 2, w22r, 64, 64, 2, 4);
  PW(H31 + 2, w31r, 64, 64, 2, 4);
  PW(H32 + 2, w32r, 64, 64, 2, 4);
  PW(V1, wv1, 68, 64, 3, 4);
  PW(V2, wv2, 80, 64, 3, 4);
  PW(FC1, wfc1, 192, 64, 6, 4);
  PW(FC2, wfc2, 64, 32, 2, 2);
  PW(FC3, wfc3, 32, 10, 1, 1);

  // ---- level 1: three NNConv layers
  FILL(agg, (size_t)kN1 * 64);
  nnconv_msg_wmma<4, 16, 4, 96><<<MSG_BLK(kE1), 256, 0, stream>>>(in_x, ei1, ea1, w11e, agg, kE1);
  LIN4(in_x, 16, w11r, (const float*)d_in[H11 + 3], agg, 64, bufA, 64, kN1, 16, 32, 64, 1);

  FILL(agg, (size_t)kN1 * 64);
  nnconv_msg_wmma<4, 64, 6, 320><<<MSG_BLK(kE1), 256, 0, stream>>>(bufA, ei1, ea1, w12e, agg, kE1);
  LIN4(bufA, 64, w12r, (const float*)d_in[H12 + 3], agg, 64, bufB, 64, kN1, 64, 64, 64, 1);

  FILL(agg, (size_t)kN1 * 64);
  nnconv_msg_wmma<4, 64, 6, 320><<<MSG_BLK(kE1), 256, 0, stream>>>(bufB, ei1, ea1, w13e, agg, kE1);
  LIN4(bufB, 64, w13r, (const float*)d_in[H13 + 3], agg, 64, bufA, 64, kN1, 64, 64, 64, 1);

  // x_1 = segment_sum(x, batch)
  FILL(x1g, (size_t)kG * 64);
  SCAT(bufA, 64, nullptr, batch1, x1g, 64, nullptr, kN1, 64);

  // ---- coarsen 1 -> 2
  FILL(cat2, (size_t)kN2 * 68);
  FILL(cnt2, kN2);
  SCAT(bufA, 64, as12, as12 + kN1, cat2, 68, cnt2, kN1, 64);
  DIV(cat2, 68, cnt2, kN2, 64);
  CPY(aedge2, 4, cat2, 68, 64, kN2, 4);
  LIN4(cat2, 68, wv1, (const float*)d_in[V1 + 1], nullptr, 0, x2a, 64, kN2, 68, 96, 64, 0);

  FILL(agg, (size_t)kN2 * 64);
  nnconv_msg_wmma<16, 64, 6, 1088><<<MSG_BLK(kE2), 256, 0, stream>>>(x2a, ei2, ea2, w21e, agg, kE2);
  LIN4(x2a, 64, w21r, (const float*)d_in[H21 + 3], agg, 64, x2b, 64, kN2, 64, 64, 64, 1);

  FILL(agg, (size_t)kN2 * 64);
  nnconv_msg_wmma<16, 64, 6, 1088><<<MSG_BLK(kE2), 256, 0, stream>>>(x2b, ei2, ea2, w22e, agg, kE2);
  LIN4(x2b, 64, w22r, (const float*)d_in[H22 + 3], agg, 64, x2a, 64, kN2, 64, 64, 64, 1);

  FILL(x2g, (size_t)kG * 64);
  FILL(cntG, kG);
  SCAT(x2a, 64, nullptr, batch2, x2g, 64, cntG, kN2, 64);
  DIV(x2g, 64, cntG, kG, 64);

  // ---- coarsen 2 -> 3
  FILL(cat3, (size_t)kN3 * 80);
  FILL(cnt3, kN3);
  SCAT(x2a, 64, as23, as23 + kN2, cat3, 80, cnt3, kN2, 64);
  DIV(cat3, 80, cnt3, kN3, 64);
  CPY(aedge3, 16, cat3, 80, 64, kN3, 16);
  LIN4(cat3, 80, wv2, (const float*)d_in[V2 + 1], nullptr, 0, x3a, 64, kN3, 80, 96, 64, 0);

  // avg_pool applied again with assi_23 (as in reference)
  FILL(x3b, (size_t)kN3 * 64);
  FILL(cnt3, kN3);
  SCAT(x3a, 64, as23, as23 + kN2, x3b, 64, cnt3, kN2, 64);
  DIV(x3b, 64, cnt3, kN3, 64);

  FILL(agg, (size_t)kN3 * 64);
  nnconv_msg_wmma<16, 64, 6, 1088><<<MSG_BLK(kE3), 256, 0, stream>>>(x3b, ei3, ea3, w31e, agg, kE3);
  LIN4(x3b, 64, w31r, (const float*)d_in[H31 + 3], agg, 64, x3a, 64, kN3, 64, 64, 64, 1);

  FILL(agg, (size_t)kN3 * 64);
  nnconv_msg_wmma<16, 64, 6, 1088><<<MSG_BLK(kE3), 256, 0, stream>>>(x3a, ei3, ea3, w32e, agg, kE3);
  LIN4(x3a, 64, w32r, (const float*)d_in[H32 + 3], agg, 64, x3b, 64, kN3, 64, 64, 64, 1);

  FILL(x3g, (size_t)kG * 64);
  FILL(cntG, kG);
  SCAT(x3b, 64, nullptr, batch3, x3g, 64, cntG, kN3, 64);
  DIV(x3g, 64, cntG, kG, 64);

  // ---- readout MLP
  CPY(x1g, 64, xcat, 192, 0, kG, 64);
  CPY(x2g, 64, xcat, 192, 64, kG, 64);
  CPY(x3g, 64, xcat, 192, 128, kG, 64);
  LIN4(xcat, 192, wfc1, (const float*)d_in[FC1 + 1], nullptr, 0, g1, 64, kG, 192, 192, 64, 1);
  wmma_linear<2><<<LIN_BLK(kG), 256, 0, stream>>>(g1, 64, wfc2, (const float*)d_in[FC2 + 1],
                                                  nullptr, 0, g2, 32, kG, 64, 64, 32, 1);
  wmma_linear<1><<<LIN_BLK(kG), 256, 0, stream>>>(g2, 32, wfc3, (const float*)d_in[FC3 + 1],
                                                  nullptr, 0, g3, 16, kG, 32, 32, 10, 0);
  log_softmax10<<<divup(kG, 256), 256, 0, stream>>>(g3, 16, out, kG);
}
